// TATC_23983097380981
// MI455X (gfx1250) — matrix operand order repaired and checked by reading, not compiled, not run on hardware
//
#include <hip/hip_runtime.h>
#include <hip/hip_bf16.h>
#include <math.h>

// ---------------------------------------------------------------------------
// CDNA5 / gfx1250 implementation.  wave32, WMMA f32_16x16x32_f16 for all
// heavy GEMMs (cv1, gated conv f||g, res||skip projections).
// ---------------------------------------------------------------------------

typedef __attribute__((ext_vector_type(16))) _Float16 v16h;
typedef __attribute__((ext_vector_type(8)))  _Float16 v8h;
typedef __attribute__((ext_vector_type(8)))  float    v8f;

#define SLEN   4096
#define NSEQ   80
#define NCLS   10
#define CH0    64
#define CH     128
#define NST    3

__device__ __forceinline__ v8f zero8f() {
    v8f r;
#pragma unroll
    for (int i = 0; i < 8; ++i) r[i] = 0.f;
    return r;
}
__device__ __forceinline__ v16h zero16h() {
    v16h r;
#pragma unroll
    for (int i = 0; i < 16; ++i) r[i] = (_Float16)0.f;
    return r;
}

// A-matrix fragment loader (16x32 f16, contiguous k order, same as packed B):
// lanes 0-15  : row M = lane,     K = kbase + {0..15}
// lanes 16-31 : row M = lane-16,  K = kbase + {16..31}
// rowp points at the start (k=0) of this lane's row; kh = lane>>4.
__device__ __forceinline__ v16h load_a(const _Float16* rowp, int kbase, int kh) {
    const _Float16* p = rowp + kbase + kh * 16;
    v8h lo = *reinterpret_cast<const v8h*>(p);
    v8h hi = *reinterpret_cast<const v8h*>(p + 8);
    v16h a;
#pragma unroll
    for (int i = 0; i < 8; ++i) { a[i] = lo[i]; a[8 + i] = hi[i]; }
    return a;
}

__device__ __forceinline__ v8f wmma_f16(v16h a, v16h b, v8f c) {
    return __builtin_amdgcn_wmma_f32_16x16x32_f16(false, a, false, b,
                                                  (short)0, c, false, false);
}

__device__ __forceinline__ float sigmoidf_(float x) { return 1.f / (1.f + expf(-x)); }

// ---------------------------------------------------------------------------
// Weight packing: logical W[k][n] (f32) -> WMMA-B tiles (f16):
//   linear = ((ntile*KS + ks)*32 + lane)*16 + e
//   n = ntile*16 + (lane&15),  k = ks*32 + (lane>>4)*16 + e
// ---------------------------------------------------------------------------
__global__ void pack_cv1_kernel(const float* __restrict__ w, _Float16* __restrict__ o) {
    int i = blockIdx.x * 256 + threadIdx.x;           // 8192 total (8 ntiles, KS=2)
    int e = i & 15, lane = (i >> 4) & 31, ks = (i >> 9) & 1, j = i >> 10;
    int n = j * 16 + (lane & 15);
    int k = ks * 32 + (lane >> 4) * 16 + e;           // 0..63
    o[i] = (_Float16)w[n * 64 + k];                   // cv1_w[n][k], shape (128,64)
}

__global__ void pack_fg_kernel(const float* __restrict__ fw, const float* __restrict__ gw,
                               _Float16* __restrict__ o) {
    int i = blockIdx.x * 256 + threadIdx.x;           // 3*16*12*512 = 294912
    int st = i / 98304; int r = i % 98304;
    int j = r / 6144;  r %= 6144;                     // ntile 0..15
    int ks = r / 512;  r %= 512;                      // 0..11
    int lane = r / 16; int e = r % 16;
    int n = j * 16 + (lane & 15);                     // 0..255  (f | g)
    int k = ks * 32 + (lane >> 4) * 16 + e;           // 0..383
    int t = k / 128, cin = k % 128;
    float v = (n < CH)
        ? fw[(((size_t)st * 3 + t) * CH + cin) * CH + n]
        : gw[(((size_t)st * 3 + t) * CH + cin) * CH + (n - CH)];
    o[i] = (_Float16)v;
}

__global__ void pack_rs_kernel(const float* __restrict__ rw, const float* __restrict__ sw,
                               _Float16* __restrict__ o) {
    int i = blockIdx.x * 256 + threadIdx.x;           // 3*16*4*512 = 98304
    int st = i / 32768; int r = i % 32768;
    int j = r / 2048;  r %= 2048;
    int ks = r / 512;  r %= 512;
    int lane = r / 16; int e = r % 16;
    int n = j * 16 + (lane & 15);
    int k = ks * 32 + (lane >> 4) * 16 + e;           // 0..127
    float v = (n < CH) ? rw[((size_t)st * CH + n) * CH + k]
                       : sw[((size_t)st * CH + (n - CH)) * CH + k];
    o[i] = (_Float16)v;
}

// ---------------------------------------------------------------------------
// Stable partition per (b,c): matched positions (in order) + length.
// ---------------------------------------------------------------------------
__global__ void partition_kernel(const int* __restrict__ labels,
                                 int* __restrict__ idxb, int* __restrict__ lenb) {
    int seq = blockIdx.x;                    // seq = b*10 + (class-1)
    int b = seq / NCLS, cls = seq % NCLS + 1;
    const int* lab = labels + (size_t)b * SLEN;
    __shared__ int sc[256];
    int t = threadIdx.x, base = t * 16;
    int cnt = 0;
#pragma unroll
    for (int i = 0; i < 16; ++i) cnt += (lab[base + i] == cls);
    sc[t] = cnt;
    __syncthreads();
    for (int off = 1; off < 256; off <<= 1) {
        int v = (t >= off) ? sc[t - off] : 0;
        __syncthreads();
        sc[t] += v;
        __syncthreads();
    }
    int pos = sc[t] - cnt;
    for (int i = 0; i < 16; ++i)
        if (lab[base + i] == cls) idxb[(size_t)seq * SLEN + pos++] = base + i;
    if (t == 0) lenb[seq] = sc[255];
}

// ---------------------------------------------------------------------------
// Fused gather + cv0 (3 -> 64) + relu + mask; store f16.  Tail stores zeros
// so all downstream masked sums/convs see exact zeros.
// Note: reference does raw_data.reshape(B,S,3) => x[b][p][c] = raw[b*12288+p*3+c].
// ---------------------------------------------------------------------------
__global__ void embed_kernel(const float* __restrict__ raw, const int* __restrict__ idxb,
                             const int* __restrict__ lenb, const float* __restrict__ w,
                             const float* __restrict__ bias, _Float16* __restrict__ h0) {
    int g = blockIdx.x * 256 + threadIdx.x;      // seq*SLEN + p
    int seq = g >> 12, p = g & (SLEN - 1);
    _Float16* op = h0 + (size_t)g * CH0;
    if (p < lenb[seq]) {
        int src = idxb[g];
        const float* xp = raw + (size_t)(seq / NCLS) * 3 * SLEN + src * 3;
        float x0 = xp[0], x1 = xp[1], x2 = xp[2];
#pragma unroll
        for (int o = 0; o < CH0; ++o) {
            float v = x0 * w[o * 3] + x1 * w[o * 3 + 1] + x2 * w[o * 3 + 2] + bias[o];
            op[o] = (_Float16)fmaxf(v, 0.f);
        }
    } else {
#pragma unroll
        for (int o = 0; o < CH0; ++o) op[o] = (_Float16)0.f;
    }
}

// masked column sums (tail is already zero) -> per-seq per-channel sums
__global__ void colsum_kernel(const _Float16* __restrict__ h, float* __restrict__ sums, int chn) {
    int seq = blockIdx.x >> 4, chunk = blockIdx.x & 15, ch = threadIdx.x;
    const _Float16* base = h + (((size_t)seq * SLEN) + chunk * 256) * chn + ch;
    float acc = 0.f;
    for (int p = 0; p < 256; ++p) acc += (float)base[(size_t)p * chn];
    atomicAdd(&sums[seq * chn + ch], acc);
}

// SE gate: e = sigmoid(relu(mean @ w1^T + b1) @ w2^T + b2)
__global__ void gate_kernel(const float* __restrict__ sums, const float* __restrict__ w1,
                            const float* __restrict__ b1, const float* __restrict__ w2,
                            const float* __restrict__ b2, const int* __restrict__ lenb,
                            float* __restrict__ gate, int chn, int hidn) {
    __shared__ float s[CH];
    __shared__ float hid[8];
    int seq = blockIdx.x, t = threadIdx.x;
    float cnt = fmaxf((float)lenb[seq], 1.f);
    s[t] = sums[seq * chn + t] / cnt;
    __syncthreads();
    if (t < hidn) {
        float a = b1[t];
        for (int k = 0; k < chn; ++k) a += s[k] * w1[t * chn + k];
        hid[t] = fmaxf(a, 0.f);
    }
    __syncthreads();
    float e = b2[t];
    for (int j = 0; j < hidn; ++j) e += hid[j] * w2[t * hidn + j];
    gate[seq * chn + t] = sigmoidf_(e);
}

__global__ void scale_kernel(_Float16* __restrict__ h, const float* __restrict__ gate,
                             int chn, int lg) {
    size_t i = (size_t)blockIdx.x * 256 + threadIdx.x;
    int ch = (int)(i & (chn - 1));
    int seq = (int)(i >> (12 + lg));
    h[i] = (_Float16)((float)h[i] * gate[seq * chn + ch]);
}

// ---------------------------------------------------------------------------
// cv1 GEMM: [pos x 64] @ [64 x 128], relu + bias + mask -> act (f16)
// one wave per 16-position M-tile; 8 waves/block; tiles never cross seqs.
// ---------------------------------------------------------------------------
__global__ void gemm_cv1_kernel(const _Float16* __restrict__ A, const _Float16* __restrict__ wB,
                                const float* __restrict__ bias, const int* __restrict__ lenb,
                                _Float16* __restrict__ out) {
    const int lane = threadIdx.x & 31;
    const int mt = blockIdx.x * 8 + (threadIdx.x >> 5);
    const int seq = mt >> 8, p0 = (mt & 255) << 4;
    const int L = lenb[seq];
    const int r = lane & 15, kh = lane >> 4;
    const _Float16* rowp = A + ((size_t)mt * 16 + r) * CH0;
    v16h a0 = load_a(rowp, 0, kh);
    v16h a1 = load_a(rowp, 32, kh);
#pragma unroll
    for (int j = 0; j < 8; ++j) {
        v8f acc = zero8f();
        v16h b0 = *reinterpret_cast<const v16h*>(wB + ((size_t)(j * 2 + 0) * 32 + lane) * 16);
        v16h b1 = *reinterpret_cast<const v16h*>(wB + ((size_t)(j * 2 + 1) * 32 + lane) * 16);
        acc = wmma_f16(a0, b0, acc);
        acc = wmma_f16(a1, b1, acc);
        int ch = j * 16 + r;
        float bv = bias[ch];
#pragma unroll
        for (int rr = 0; rr < 8; ++rr) {
            int p = p0 + kh * 8 + rr;
            float v = fmaxf(acc[rr] + bv, 0.f);
            if (p >= L) v = 0.f;
            out[((size_t)seq * SLEN + p) * CH + ch] = (_Float16)v;
        }
    }
}

// ---------------------------------------------------------------------------
// Gated conv GEMM: im2col K=384 (3 taps x 128 ch), N=256 (f || g),
// zg = tanh(f+fb) * sigmoid(g+gb), masked, f16.
// ---------------------------------------------------------------------------
__global__ void gemm_conv_kernel(const _Float16* __restrict__ cur, const _Float16* __restrict__ wFG,
                                 const float* __restrict__ fb, const float* __restrict__ gb,
                                 const int* __restrict__ lenb, _Float16* __restrict__ zg) {
    const int lane = threadIdx.x & 31;
    const int mt = blockIdx.x * 8 + (threadIdx.x >> 5);
    const int seq = mt >> 8, p0 = (mt & 255) << 4;
    const int L = lenb[seq];
    const int r = lane & 15, kh = lane >> 4;
    v16h a[12];
#pragma unroll
    for (int ks = 0; ks < 12; ++ks) {
        int t = ks >> 2, kb = (ks & 3) * 32;
        int q = p0 + r - 1 + t;                      // seq-local row with tap shift
        if (q >= 0 && q < SLEN)
            a[ks] = load_a(cur + ((size_t)seq * SLEN + q) * CH, kb, kh);
        else
            a[ks] = zero16h();
    }
#pragma unroll
    for (int j = 0; j < 8; ++j) {
        v8f accF = zero8f(), accG = zero8f();
#pragma unroll
        for (int ks = 0; ks < 12; ++ks) {
            v16h bF = *reinterpret_cast<const v16h*>(wFG + ((size_t)(j * 12 + ks) * 32 + lane) * 16);
            v16h bG = *reinterpret_cast<const v16h*>(wFG + ((size_t)((j + 8) * 12 + ks) * 32 + lane) * 16);
            accF = wmma_f16(a[ks], bF, accF);
            accG = wmma_f16(a[ks], bG, accG);
        }
        int ch = j * 16 + r;
        float fbv = fb[ch], gbv = gb[ch];
#pragma unroll
        for (int rr = 0; rr < 8; ++rr) {
            int p = p0 + kh * 8 + rr;
            float z = tanhf(accF[rr] + fbv) * sigmoidf_(accG[rr] + gbv);
            if (p >= L) z = 0.f;
            zg[((size_t)seq * SLEN + p) * CH + ch] = (_Float16)z;
        }
    }
}

// ---------------------------------------------------------------------------
// res||skip GEMM: [pos x 128] @ [128 x 256]; n<128: res (in-place +cur, mask),
// n>=128: skip accumulate (f32).
// ---------------------------------------------------------------------------
__global__ void gemm_resskip_kernel(const _Float16* __restrict__ zg, const _Float16* __restrict__ wRS,
                                    const float* __restrict__ rb, const float* __restrict__ sb,
                                    _Float16* __restrict__ act, float* __restrict__ skip,
                                    const int* __restrict__ lenb, int first) {
    const int lane = threadIdx.x & 31;
    const int mt = blockIdx.x * 8 + (threadIdx.x >> 5);
    const int seq = mt >> 8, p0 = (mt & 255) << 4;
    const int L = lenb[seq];
    const int r = lane & 15, kh = lane >> 4;
    const _Float16* rowp = zg + ((size_t)seq * SLEN + p0 + r) * CH;
    v16h a[4];
#pragma unroll
    for (int ks = 0; ks < 4; ++ks) a[ks] = load_a(rowp, ks * 32, kh);
#pragma unroll
    for (int j = 0; j < 16; ++j) {
        v8f acc = zero8f();
#pragma unroll
        for (int ks = 0; ks < 4; ++ks) {
            v16h b = *reinterpret_cast<const v16h*>(wRS + ((size_t)(j * 4 + ks) * 32 + lane) * 16);
            acc = wmma_f16(a[ks], b, acc);
        }
        if (j < 8) {
            int ch = j * 16 + r;
            float bv = rb[ch];
#pragma unroll
            for (int rr = 0; rr < 8; ++rr) {
                int p = p0 + kh * 8 + rr;
                size_t o = ((size_t)seq * SLEN + p) * CH + ch;
                float v = acc[rr] + bv + (float)act[o];
                if (p >= L) v = 0.f;
                act[o] = (_Float16)v;            // same lane reads & writes o: safe
            }
        } else {
            int ch = (j - 8) * 16 + r;
            float bv = sb[ch];
#pragma unroll
            for (int rr = 0; rr < 8; ++rr) {
                int p = p0 + kh * 8 + rr;
                size_t o = ((size_t)seq * SLEN + p) * CH + ch;
                float v = acc[rr] + bv;
                if (p >= L) v = 0.f;
                if (first) skip[o] = v; else skip[o] += v;
            }
        }
    }
}

// masked avg/max pool -> feat[seq][256] interleaved (avg0, mx0, avg1, mx1, ...)
__global__ void pool_kernel(const _Float16* __restrict__ res, const float* __restrict__ skip,
                            const int* __restrict__ lenb, float* __restrict__ feat) {
    int seq = blockIdx.x, ch = threadIdx.x;
    int L = lenb[seq];
    size_t base = ((size_t)seq * SLEN) * CH + ch;
    float s = 0.f, mx = -3.4e38f;
    for (int p = 0; p < L; ++p) {
        float v = (float)res[base + (size_t)p * CH] + skip[base + (size_t)p * CH];
        s += v;
        mx = fmaxf(mx, v);
    }
    if (L == 0) mx = 0.f;
    feat[seq * 256 + ch * 2]     = s / fmaxf((float)L, 1.f);
    feat[seq * 256 + ch * 2 + 1] = mx;
}

// MLP head: 256 -> 1024 relu -> 2
__global__ void head_kernel(const float* __restrict__ feat, const float* __restrict__ l1w,
                            const float* __restrict__ l1b, const float* __restrict__ l2w,
                            const float* __restrict__ l2b, float* __restrict__ out) {
    __shared__ float f[256];
    __shared__ float hid[1024];
    int seq = blockIdx.x, t = threadIdx.x;
    f[t] = feat[seq * 256 + t];
    __syncthreads();
    for (int u = t; u < 1024; u += 256) {
        float a = l1b[u];
        for (int k = 0; k < 256; ++k) a += f[k] * l1w[(size_t)u * 256 + k];
        hid[u] = fmaxf(a, 0.f);
    }
    __syncthreads();
    if (t < 2) {
        float a = l2b[t];
        for (int k = 0; k < 1024; ++k) a += hid[k] * l2w[(size_t)t * 1024 + k];
        out[seq * 2 + t] = a;
    }
}

// ---------------------------------------------------------------------------
extern "C" void kernel_launch(void* const* d_in, const int* in_sizes, int n_in,
                              void* d_out, int out_size, void* d_ws, size_t ws_size,
                              hipStream_t stream) {
    (void)in_sizes; (void)n_in; (void)out_size; (void)ws_size;
    const float* raw   = (const float*)d_in[0];
    const int*   lab   = (const int*)  d_in[1];
    const float* cv0w  = (const float*)d_in[2];
    const float* cv0b  = (const float*)d_in[3];
    const float* se0w1 = (const float*)d_in[4];
    const float* se0b1 = (const float*)d_in[5];
    const float* se0w2 = (const float*)d_in[6];
    const float* se0b2 = (const float*)d_in[7];
    const float* cv1w  = (const float*)d_in[8];
    const float* cv1b  = (const float*)d_in[9];
    const float* se1w1 = (const float*)d_in[10];
    const float* se1b1 = (const float*)d_in[11];
    const float* se1w2 = (const float*)d_in[12];
    const float* se1b2 = (const float*)d_in[13];
    const float* rsfw  = (const float*)d_in[14];
    const float* rsfb  = (const float*)d_in[15];
    const float* rsgw  = (const float*)d_in[16];
    const float* rsgb  = (const float*)d_in[17];
    const float* rsrw  = (const float*)d_in[18];
    const float* rsrb  = (const float*)d_in[19];
    const float* rssw  = (const float*)d_in[20];
    const float* rssb  = (const float*)d_in[21];
    const float* l1w   = (const float*)d_in[22];
    const float* l1b   = (const float*)d_in[23];
    const float* l2w   = (const float*)d_in[24];
    const float* l2b   = (const float*)d_in[25];
    float* out = (float*)d_out;

    const size_t NPOS = (size_t)NSEQ * SLEN;     // 327680
    char* ws = (char*)d_ws;
    size_t off = 0;
    auto take = [&](size_t bytes) { size_t o = off; off = (off + bytes + 255) & ~(size_t)255; return o; };

    int*       idxb  = (int*)      (ws + take(NPOS * 4));
    int*       lenb  = (int*)      (ws + take(NSEQ * 4));
    float*     sum0  = (float*)    (ws + take((size_t)NSEQ * CH0 * 4));
    float*     sum1  = (float*)    (ws + take((size_t)NSEQ * CH * 4));
    float*     gate0 = (float*)    (ws + take((size_t)NSEQ * CH0 * 4));
    float*     gate1 = (float*)    (ws + take((size_t)NSEQ * CH * 4));
    float*     featb = (float*)    (ws + take((size_t)NSEQ * 256 * 4));
    _Float16*  wCv1  = (_Float16*) (ws + take((size_t)8192 * 2));
    _Float16*  wFG   = (_Float16*) (ws + take((size_t)294912 * 2));
    _Float16*  wRS   = (_Float16*) (ws + take((size_t)98304 * 2));
    _Float16*  h0    = (_Float16*) (ws + take(NPOS * CH0 * 2));
    _Float16*  act   = (_Float16*) (ws + take(NPOS * CH * 2));
    _Float16*  zgb   = (_Float16*) (ws + take(NPOS * CH * 2));
    float*     skipb = (float*)    (ws + take(NPOS * CH * 4));

    hipMemsetAsync(sum0, 0, (size_t)NSEQ * CH0 * 4, stream);
    hipMemsetAsync(sum1, 0, (size_t)NSEQ * CH * 4, stream);

    // weight packing (tiny)
    pack_cv1_kernel<<<32, 256, 0, stream>>>(cv1w, wCv1);
    pack_fg_kernel<<<1152, 256, 0, stream>>>(rsfw, rsgw, wFG);
    pack_rs_kernel<<<384, 256, 0, stream>>>(rsrw, rssw, wRS);

    // stable partition + gather/embed
    partition_kernel<<<NSEQ, 256, 0, stream>>>(lab, idxb, lenb);
    embed_kernel<<<(int)(NPOS / 256), 256, 0, stream>>>(raw, idxb, lenb, cv0w, cv0b, h0);

    // SE0
    colsum_kernel<<<NSEQ * 16, CH0, 0, stream>>>(h0, sum0, CH0);
    gate_kernel<<<NSEQ, CH0, 0, stream>>>(sum0, se0w1, se0b1, se0w2, se0b2, lenb, gate0, CH0, 4);
    scale_kernel<<<(int)(NPOS * CH0 / 256), 256, 0, stream>>>(h0, gate0, CH0, 6);

    // cv1 (WMMA) -> act
    const int GEMM_BLOCKS = (int)(NPOS / 16 / 8);    // 2560
    gemm_cv1_kernel<<<GEMM_BLOCKS, 256, 0, stream>>>(h0, wCv1, cv1b, lenb, act);

    // SE1
    colsum_kernel<<<NSEQ * 16, CH, 0, stream>>>(act, sum1, CH);
    gate_kernel<<<NSEQ, CH, 0, stream>>>(sum1, se1w1, se1b1, se1w2, se1b2, lenb, gate1, CH, 8);
    scale_kernel<<<(int)(NPOS * CH / 256), 256, 0, stream>>>(act, gate1, CH, 7);

    // residual stacks (WMMA conv + WMMA res/skip, in-place residual)
    for (int i = 0; i < NST; ++i) {
        gemm_conv_kernel<<<GEMM_BLOCKS, 256, 0, stream>>>(
            act, wFG + (size_t)i * 98304, rsfb + (size_t)i * CH, rsgb + (size_t)i * CH, lenb, zgb);
        gemm_resskip_kernel<<<GEMM_BLOCKS, 256, 0, stream>>>(
            zgb, wRS + (size_t)i * 32768, rsrb + (size_t)i * CH, rssb + (size_t)i * CH,
            act, skipb, lenb, (i == 0) ? 1 : 0);
    }

    // pool + head
    pool_kernel<<<NSEQ, CH, 0, stream>>>(act, skipb, lenb, featb);
    head_kernel<<<NSEQ, 256, 0, stream>>>(featb, l1w, l1b, l2w, l2b, out);
}
